// SwinTransformerBlock_12180527251720
// MI455X (gfx1250) — matrix operand order repaired and checked by reading, not compiled, not run on hardware
//
#include <hip/hip_runtime.h>
#include <hip/hip_bf16.h>
#include <stdint.h>

// ---------------------------------------------------------------------------
// Swin Transformer block on gfx1250 (MI455X), f16 WMMA everywhere.
// B=8, H=W=128, C=192, WS=8, SS=4, NH=6, HD=32, N=64, nW=256, B_=2048,
// Mtok = B*H*W = 131072 tokens.
// ---------------------------------------------------------------------------

typedef __attribute__((ext_vector_type(16))) _Float16 v16h;
typedef __attribute__((ext_vector_type(8)))  float    v8f;

#define MTOK   131072
#define CDIM   192
#define SCALE_Q 0.17677669529663687f   // 32^-0.5

// ---------------------------------------------------------------------------
// Weight convert: w[K][N] fp32  ->  wt[N][K] f16   (transpose + downconvert)
// ---------------------------------------------------------------------------
__global__ void wt_convert_kernel(const float* __restrict__ w,
                                  _Float16* __restrict__ wt, int K, int N) {
    long i = (long)blockIdx.x * blockDim.x + threadIdx.x;
    long total = (long)K * N;
    if (i >= total) return;
    int k = (int)(i / N), n = (int)(i % N);
    wt[(long)n * K + k] = (_Float16)w[i];
}

// ---------------------------------------------------------------------------
// LayerNorm over C=192, fused (optionally) with roll(-4,-4)+window partition.
// One block per output token, 192 threads (6 waves), f16 output.
// ---------------------------------------------------------------------------
template<int SHIFTED>
__global__ __launch_bounds__(192) void ln_kernel(const float* __restrict__ in,
                                                 const float* __restrict__ scale,
                                                 const float* __restrict__ bias,
                                                 _Float16* __restrict__ out) {
    __shared__ float red[8];
    int t = blockIdx.x;      // output token (window order if SHIFTED)
    int c = threadIdx.x;     // channel 0..191
    long src;
    if (SHIFTED) {
        int b  = t >> 14;           // /(256*64)
        int wi = (t >> 6) & 255;    // window in batch
        int nn = t & 63;            // token in window
        int hs = ((wi >> 4) << 3) + (nn >> 3);
        int ws = ((wi & 15) << 3) + (nn & 7);
        int h  = (hs + 4) & 127;    // roll by -SS => read shifted+SS
        int w  = (ws + 4) & 127;
        src = ((long)b << 14) + (h << 7) + w;
    } else {
        src = t;
    }
    float v = in[src * CDIM + c];
    // mean
    float s = v;
    #pragma unroll
    for (int o = 16; o; o >>= 1) s += __shfl_xor(s, o, 32);
    int wv = c >> 5;
    if ((c & 31) == 0) red[wv] = s;
    __syncthreads();
    float mean = (red[0] + red[1] + red[2] + red[3] + red[4] + red[5]) * (1.0f / 192.0f);
    float d = v - mean;
    float s2 = d * d;
    #pragma unroll
    for (int o = 16; o; o >>= 1) s2 += __shfl_xor(s2, o, 32);
    __syncthreads();
    if ((c & 31) == 0) red[wv] = s2;
    __syncthreads();
    float var = (red[0] + red[1] + red[2] + red[3] + red[4] + red[5]) * (1.0f / 192.0f);
    float y = d * rsqrtf(var + 1e-5f) * scale[c] + bias[c];
    out[(long)t * CDIM + c] = (_Float16)y;
}

// ---------------------------------------------------------------------------
// Generic f16 WMMA GEMM:  out = epilogue(A[M,K] @ Wt[N,K]^T + bias)
// Block tile 64x64, 256 threads = 8 waves, each wave: 1 m-tile x 2 n-tiles.
// Fragments (16-bit A 16x32 / B 32x16) use contiguous k order: k = 16*hi + s, same in A and B.
// ---------------------------------------------------------------------------
enum { EP_F16BIAS = 0, EP_PROJ = 1, EP_GELU = 2, EP_FC2 = 3 };

template<int EPI>
__global__ __launch_bounds__(256) void gemm_wmma_kernel(
    const _Float16* __restrict__ A,   // [M][K] f16, row-major
    const _Float16* __restrict__ Wt,  // [N][K] f16 (pre-transposed weights)
    const float*    __restrict__ bias,
    const float*    __restrict__ resid,
    void*           __restrict__ out,
    int M, int N, int K) {
    __shared__ __align__(16) _Float16 As[64 * 32];
    __shared__ __align__(16) _Float16 Bs[64 * 32];
    const int tid  = threadIdx.x;
    const int m0   = blockIdx.y * 64;
    const int n0   = blockIdx.x * 64;
    const int wv   = tid >> 5;
    const int lane = tid & 31;
    const int lr   = lane & 15;
    const int hi   = lane >> 4;
    const int r    = wv & 3;              // m-tile
    const int c0   = (wv >> 2) * 2;       // first n-tile of this wave
    const int ldr  = tid >> 2;            // 0..63 row for cooperative loads
    const int ldc  = (tid & 3) * 8;       // half offset (0/8/16/24)

    v8f acc0 = {}; v8f acc1 = {};
    const uint32_t* As32 = (const uint32_t*)As;
    const uint32_t* Bs32 = (const uint32_t*)Bs;
    const int mrow  = r * 16 + lr;
    const int nrow0 = c0 * 16 + lr;
    const int nrow1 = nrow0 + 16;

    for (int kb = 0; kb < K; kb += 32) {
        __syncthreads();
        *(uint4*)&As[ldr * 32 + ldc] = *(const uint4*)&A [(long)(m0 + ldr) * K + kb + ldc];
        *(uint4*)&Bs[ldr * 32 + ldc] = *(const uint4*)&Wt[(long)(n0 + ldr) * K + kb + ldc];
        __syncthreads();
        union { v16h v; uint32_t u[8]; } a, b0, b1;
        #pragma unroll
        for (int j = 0; j < 8; ++j) {   // A: 16x32 f16, contiguous k = hi*16 + s
            int kp = hi * 16 + 2 * j;
            a.u[j] = As32[(mrow * 32 + kp) >> 1];
        }
        #pragma unroll
        for (int j = 0; j < 8; ++j) {   // B: 32x16 f16 layout, Bs is n-major
            int kp = hi * 16 + 2 * j;
            b0.u[j] = Bs32[(nrow0 * 32 + kp) >> 1];
            b1.u[j] = Bs32[(nrow1 * 32 + kp) >> 1];
        }
        acc0 = __builtin_amdgcn_wmma_f32_16x16x32_f16(false, a.v, false, b0.v, (short)0, acc0, false, false);
        acc1 = __builtin_amdgcn_wmma_f32_16x16x32_f16(false, a.v, false, b1.v, (short)0, acc1, false, false);
    }

    #pragma unroll
    for (int t2 = 0; t2 < 2; ++t2) {
        v8f acc; if (t2) acc = acc1; else acc = acc0;
        int ncol = n0 + (c0 + t2) * 16 + lr;
        float bn = bias[ncol];
        #pragma unroll
        for (int i = 0; i < 8; ++i) {   // C/D layout: VGPR i -> M = i + hi*8
            int mr = m0 + r * 16 + i + hi * 8;
            float val = acc[i] + bn;
            if (EPI == EP_F16BIAS) {
                ((_Float16*)out)[(long)mr * N + ncol] = (_Float16)val;
            } else if (EPI == EP_GELU) {
                float xx = val;
                float g = 0.5f * xx * (1.0f + tanhf(0.7978845608f * (xx + 0.044715f * xx * xx * xx)));
                ((_Float16*)out)[(long)mr * N + ncol] = (_Float16)g;
            } else if (EPI == EP_FC2) {
                ((float*)out)[(long)mr * N + ncol] = val + resid[(long)mr * N + ncol];
            } else { // EP_PROJ: window reverse + roll(+4,+4) + residual
                int win = mr >> 6, nn = mr & 63;
                int b   = win >> 8, wi = win & 255;
                int hs  = ((wi >> 4) << 3) + (nn >> 3);
                int ws  = ((wi & 15) << 3) + (nn & 7);
                int h   = (hs + 4) & 127, w = (ws + 4) & 127;
                long dest = ((long)b << 14) + (h << 7) + w;
                ((float*)out)[dest * CDIM + ncol] = val + resid[dest * CDIM + ncol];
            }
        }
    }
}

// ---------------------------------------------------------------------------
// Windowed attention, one block per (window, head). 128 threads = 4 waves.
// S = (q @ k^T)*SCALE + rpb[rpi] + shiftmask ; softmax ; out = P @ v.
// ---------------------------------------------------------------------------
__global__ __launch_bounds__(128) void attn_kernel(const _Float16* __restrict__ qkv,
                                                   const float* __restrict__ rpb,
                                                   _Float16* __restrict__ out) {
    __shared__ __align__(16) _Float16 Qs[64 * 32];
    __shared__ __align__(16) _Float16 Ks[64 * 32];
    __shared__ __align__(16) _Float16 Ps[64 * 64];
    __shared__ __align__(16) _Float16 Vs[32 * 64];   // Vs[col][tok]
    __shared__ float Sf[64 * 64];

    int bid  = blockIdx.x;
    int win  = bid / 6, head = bid % 6;
    int row0 = win * 64;
    int qoff = head * 32, koff = 192 + head * 32, voff = 384 + head * 32;
    int tid  = threadIdx.x;

    for (int ch = tid; ch < 256; ch += 128) {
        int rr = ch >> 2, cc = (ch & 3) * 8;
        *(uint4*)&Qs[rr * 32 + cc] = *(const uint4*)&qkv[(long)(row0 + rr) * 576 + qoff + cc];
        *(uint4*)&Ks[rr * 32 + cc] = *(const uint4*)&qkv[(long)(row0 + rr) * 576 + koff + cc];
    }
    for (int e = tid; e < 2048; e += 128) {
        int tok = e >> 5, col = e & 31;
        Vs[col * 64 + tok] = qkv[(long)(row0 + tok) * 576 + voff + col];
    }
    __syncthreads();

    int wv = tid >> 5, lane = tid & 31, lr = lane & 15, hi = lane >> 4;
    const uint32_t* Qs32 = (const uint32_t*)Qs;
    const uint32_t* Ks32 = (const uint32_t*)Ks;
    int mrow = wv * 16 + lr;

    union { v16h v; uint32_t u[8]; } a;
    #pragma unroll
    for (int j = 0; j < 8; ++j) {
        int kp = hi * 16 + 2 * j;
        a.u[j] = Qs32[(mrow * 32 + kp) >> 1];
    }

    int wrow8 = (((win & 255) >> 4) << 3);   // window origin in image coords
    int wcol8 = ((win & 255) & 15) << 3;

    #pragma unroll
    for (int ct = 0; ct < 4; ++ct) {
        union { v16h v; uint32_t u[8]; } b;
        int nrow = ct * 16 + lr;
        #pragma unroll
        for (int j = 0; j < 8; ++j)
            b.u[j] = Ks32[(nrow * 32 + hi * 16 + 2 * j) >> 1];
        v8f s = {};
        s = __builtin_amdgcn_wmma_f32_16x16x32_f16(false, a.v, false, b.v, (short)0, s, false, false);
        #pragma unroll
        for (int i = 0; i < 8; ++i) {
            int m = wv * 16 + i + hi * 8;    // query token
            int n = nrow;                    // key token
            int dy = (m >> 3) - (n >> 3) + 7;
            int dx = (m & 7)  - (n & 7)  + 7;
            float bi = rpb[(dy * 15 + dx) * 6 + head];
            int hm = wrow8 + (m >> 3), wm = wcol8 + (m & 7);
            int hn = wrow8 + (n >> 3), wn = wcol8 + (n & 7);
            int cm = (hm < 120 ? 0 : (hm < 124 ? 1 : 2)) * 3 + (wm < 120 ? 0 : (wm < 124 ? 1 : 2));
            int cn = (hn < 120 ? 0 : (hn < 124 ? 1 : 2)) * 3 + (wn < 120 ? 0 : (wn < 124 ? 1 : 2));
            float msk = (cm == cn) ? 0.0f : -100.0f;
            Sf[m * 64 + n] = s[i] * SCALE_Q + bi + msk;
        }
    }
    __syncthreads();

    { // softmax over rows of 64: two threads per row (paired lanes)
        int rrow = tid >> 1, seg = tid & 1;
        float* rp = &Sf[rrow * 64 + seg * 32];
        float mx = -1e30f;
        #pragma unroll
        for (int j = 0; j < 32; ++j) mx = fmaxf(mx, rp[j]);
        mx = fmaxf(mx, __shfl_xor(mx, 1, 32));
        float ev[32]; float sum = 0.0f;
        #pragma unroll
        for (int j = 0; j < 32; ++j) { ev[j] = __expf(rp[j] - mx); sum += ev[j]; }
        sum += __shfl_xor(sum, 1, 32);
        float inv = 1.0f / sum;
        #pragma unroll
        for (int j = 0; j < 32; ++j)
            Ps[rrow * 64 + seg * 32 + j] = (_Float16)(ev[j] * inv);
    }
    __syncthreads();

    // out(64x32) = P(64x64) @ V(64x32); wave -> m-tile wv, both n-tiles
    v8f o0 = {}, o1 = {};
    const uint32_t* Ps32 = (const uint32_t*)Ps;
    const uint32_t* Vs32 = (const uint32_t*)Vs;
    #pragma unroll
    for (int ks = 0; ks < 2; ++ks) {
        union { v16h v; uint32_t u[8]; } pa, vb0, vb1;
        #pragma unroll
        for (int j = 0; j < 8; ++j) {
            int kp = hi * 16 + 2 * j + ks * 32;
            pa.u[j] = Ps32[(mrow * 64 + kp) >> 1];
        }
        #pragma unroll
        for (int j = 0; j < 8; ++j) {
            int kp = hi * 16 + 2 * j + ks * 32;
            vb0.u[j] = Vs32[(lr * 64 + kp) >> 1];
            vb1.u[j] = Vs32[((16 + lr) * 64 + kp) >> 1];
        }
        o0 = __builtin_amdgcn_wmma_f32_16x16x32_f16(false, pa.v, false, vb0.v, (short)0, o0, false, false);
        o1 = __builtin_amdgcn_wmma_f32_16x16x32_f16(false, pa.v, false, vb1.v, (short)0, o1, false, false);
    }
    #pragma unroll
    for (int i = 0; i < 8; ++i) {
        int m = wv * 16 + i + hi * 8;
        out[(long)(row0 + m) * CDIM + head * 32 + lr]      = (_Float16)o0[i];
        out[(long)(row0 + m) * CDIM + head * 32 + 16 + lr] = (_Float16)o1[i];
    }
}

// ---------------------------------------------------------------------------
extern "C" void kernel_launch(void* const* d_in, const int* in_sizes, int n_in,
                              void* d_out, int out_size, void* d_ws, size_t ws_size,
                              hipStream_t stream) {
    const float* x      = (const float*)d_in[0];
    const float* ln1_s  = (const float*)d_in[1];
    const float* ln1_b  = (const float*)d_in[2];
    const float* qkv_w  = (const float*)d_in[3];
    const float* qkv_b  = (const float*)d_in[4];
    const float* rpb    = (const float*)d_in[5];
    const float* proj_w = (const float*)d_in[6];
    const float* proj_b = (const float*)d_in[7];
    const float* ln2_s  = (const float*)d_in[8];
    const float* ln2_b  = (const float*)d_in[9];
    const float* fc1_w  = (const float*)d_in[10];
    const float* fc1_b  = (const float*)d_in[11];
    const float* fc2_w  = (const float*)d_in[12];
    const float* fc2_b  = (const float*)d_in[13];

    char* ws = (char*)d_ws;
    size_t off = 0;
    auto alloc = [&](size_t bytes) -> void* {
        void* p = ws + off;
        off = (off + bytes + 255) & ~(size_t)255;
        return p;
    };
    _Float16* qkv_wt  = (_Float16*)alloc((size_t)576 * 192 * 2);
    _Float16* proj_wt = (_Float16*)alloc((size_t)192 * 192 * 2);
    _Float16* fc1_wt  = (_Float16*)alloc((size_t)768 * 192 * 2);
    _Float16* fc2_wt  = (_Float16*)alloc((size_t)192 * 768 * 2);
    _Float16* h16     = (_Float16*)alloc((size_t)MTOK * 192 * 2);  // LN1 out, reused for LN2 out
    _Float16* bigbuf  = (_Float16*)alloc((size_t)MTOK * 768 * 2);  // QKV out (576 cols), then FC1 out (768)
    _Float16* attnout = (_Float16*)alloc((size_t)MTOK * 192 * 2);
    float*    y1      = (float*)   alloc((size_t)MTOK * 192 * 4);  // mid residual (fp32)

    // 1) weights -> f16 transposed [N][K]
    wt_convert_kernel<<<(192 * 576 + 255) / 256, 256, 0, stream>>>(qkv_w,  qkv_wt,  192, 576);
    wt_convert_kernel<<<(192 * 192 + 255) / 256, 256, 0, stream>>>(proj_w, proj_wt, 192, 192);
    wt_convert_kernel<<<(192 * 768 + 255) / 256, 256, 0, stream>>>(fc1_w,  fc1_wt,  192, 768);
    wt_convert_kernel<<<(768 * 192 + 255) / 256, 256, 0, stream>>>(fc2_w,  fc2_wt,  768, 192);

    // 2) LN1 + shift + window partition -> f16
    ln_kernel<1><<<MTOK, 192, 0, stream>>>(x, ln1_s, ln1_b, h16);

    // 3) QKV GEMM (131072 x 576 x 192) -> f16
    gemm_wmma_kernel<EP_F16BIAS><<<dim3(576 / 64, MTOK / 64), 256, 0, stream>>>(
        h16, qkv_wt, qkv_b, nullptr, (void*)bigbuf, MTOK, 576, 192);

    // 4) windowed attention (2048 windows x 6 heads)
    attn_kernel<<<2048 * 6, 128, 0, stream>>>(bigbuf, rpb, attnout);

    // 5) proj GEMM + window reverse + roll + residual -> y1 fp32
    gemm_wmma_kernel<EP_PROJ><<<dim3(192 / 64, MTOK / 64), 256, 0, stream>>>(
        attnout, proj_wt, proj_b, x, (void*)y1, MTOK, 192, 192);

    // 6) LN2 -> f16 (reuse h16)
    ln_kernel<0><<<MTOK, 192, 0, stream>>>(y1, ln2_s, ln2_b, h16);

    // 7) FC1 GEMM + GELU -> f16 (reuse bigbuf, full 768 cols)
    gemm_wmma_kernel<EP_GELU><<<dim3(768 / 64, MTOK / 64), 256, 0, stream>>>(
        h16, fc1_wt, fc1_b, nullptr, (void*)bigbuf, MTOK, 768, 192);

    // 8) FC2 GEMM + residual -> d_out fp32
    gemm_wmma_kernel<EP_FC2><<<dim3(192 / 64, MTOK / 64), 256, 0, stream>>>(
        bigbuf, fc2_wt, fc2_b, y1, d_out, MTOK, 192, 768);
}
